// ArcFace_25486335934656
// MI455X (gfx1250) — compile-verified, matrix operand order checked by reading, not run on hardware
//
#include <hip/hip_runtime.h>
#include <hip/hip_bf16.h>
#include <math.h>

typedef __attribute__((ext_vector_type(4)))  float  v4f;
typedef __attribute__((ext_vector_type(8)))  float  v8f;
typedef __attribute__((ext_vector_type(4)))  __bf16 v4bf;
typedef __attribute__((ext_vector_type(8)))  __bf16 v8bf;
typedef __attribute__((ext_vector_type(16))) __bf16 v16bf;

#define NUM_CLASSES 360232
#define DIM 512
#define BATCH 256
#define SCALE_F 64.0f
#define COS_M 0.8775825618903728f
#define SIN_M 0.479425538604203f
#define EPS_N 1e-12f

#define NCHUNK 128                                         // classes per block
#define NUM_CHUNKS ((NUM_CLASSES + NCHUNK - 1) / NCHUNK)   // 2815
#define KC 32                                              // K per bf16 WMMA
#define NKC (DIM / KC)                                     // 16 K-chunks
#define BSTR 40                                            // bf16 row stride (80B, 16B aligned, conflict-free)

// ---------------------------------------------------------------------------
// Kernel 1: row-normalize embeddings and pre-split into bf16 hi/lo, stored
// chunk-major ([chunk][row][32]) so GEMM A-staging is a pure b128 copy.
// ---------------------------------------------------------------------------
__global__ void __launch_bounds__(256) norm_split_kernel(
    const float* __restrict__ embs, __bf16* __restrict__ AhiG,
    __bf16* __restrict__ AloG) {
    const int b = blockIdx.x;
    const int t = threadIdx.x;
    float x0 = embs[b * DIM + t];
    float x1 = embs[b * DIM + t + 256];
    __shared__ float red[256];
    __shared__ float rno;
    red[t] = x0 * x0 + x1 * x1;
    __syncthreads();
    for (int off = 128; off > 0; off >>= 1) {
        if (t < off) red[t] += red[t + off];
        __syncthreads();
    }
    if (t == 0) rno = 1.0f / fmaxf(sqrtf(red[0]), EPS_N);
    __syncthreads();
    #pragma unroll
    for (int p = 0; p < 2; ++p) {
        int d = t + p * 256;
        float x = ((p == 0) ? x0 : x1) * rno;
        __bf16 h = (__bf16)x;
        __bf16 l = (__bf16)(x - (float)h);
        int idx = ((d >> 5) * BATCH + b) * KC + (d & 31);
        AhiG[idx] = h;
        AloG[idx] = l;
    }
}

// Load a 16-element bf16 fragment: K = [half*8, half*8+8) and [16+half*8, ...)
__device__ __forceinline__ v16bf load_frag(const __bf16* rowp, int half) {
    v8bf a = *(const v8bf*)(rowp + half * 8);        // 16B ds_load_b128
    v8bf b = *(const v8bf*)(rowp + 16 + half * 8);   // 16B ds_load_b128
    return __builtin_shufflevector(a, b, 0, 1, 2, 3, 4, 5, 6, 7,
                                         8, 9, 10, 11, 12, 13, 14, 15);
}

// ---------------------------------------------------------------------------
// Kernel 2: fused cosine-GEMM via 3xBF16 split-precision WMMA
//           + inline weight-row norms + margin + per-(row,chunk) LSE partials.
// Block = 256 threads = 8 waves. Tile: M=256 (all batch) x N=128 classes.
// ---------------------------------------------------------------------------
__global__ void __launch_bounds__(256) arcface_gemm_kernel(
    const __bf16* __restrict__ AhiG, const __bf16* __restrict__ AloG,
    const float* __restrict__ W, const int* __restrict__ idtys,
    float* __restrict__ pmax, float* __restrict__ psum,
    float* __restrict__ tgtl) {
    extern __shared__ char smem[];
    __bf16* Ahi   = (__bf16*)smem;                   // 256*40 bf16 = 20480 B
    __bf16* Alo   = (__bf16*)(smem + 20480);         // 20480 B
    __bf16* Bhi   = (__bf16*)(smem + 40960);         // 128*40 bf16 = 10240 B
    __bf16* Blo   = (__bf16*)(smem + 51200);         // 10240 B
    float*  wnorm = (float*)(smem + 61440);          // 128 f32
    int*    sIdty = (int*)(smem + 61952);            // 256 i32
    float*  redmax = (float*)smem;                   // alias Ahi (post-GEMM)
    float*  redsum = (float*)smem + 2048;            // alias Ahi

    const int tid   = threadIdx.x;
    const int wave  = tid >> 5;
    const int lane  = tid & 31;
    const int l15   = lane & 15;
    const int half  = lane >> 4;
    const int chunk = blockIdx.x;
    const int cbase = chunk * NCHUNK;

    sIdty[tid] = idtys[tid];
    if (tid < NCHUNK) wnorm[tid] = 0.0f;

    v8f acc[16];
    #pragma unroll
    for (int i = 0; i < 16; ++i)
        #pragma unroll
        for (int j = 0; j < 8; ++j) acc[i][j] = 0.0f;

    float ssq[4] = {0.f, 0.f, 0.f, 0.f};   // partial sumsq for 4 fixed B rows

    for (int kc = 0; kc < DIM; kc += KC) {
        __syncthreads();
        // Stage A chunk: pure b128 copy of pre-split bf16 (no VALU).
        // 256 rows x 32 bf16 = 1024 v8bf per matrix; 4 per thread each.
        {
            const __bf16* srch = AhiG + (size_t)(kc >> 5) * (BATCH * KC);
            const __bf16* srcl = AloG + (size_t)(kc >> 5) * (BATCH * KC);
            #pragma unroll
            for (int j = 0; j < 4; ++j) {
                int i = tid + j * 256;
                int row = i >> 2, col8 = i & 3;
                v8bf h = *(const v8bf*)(srch + row * KC + col8 * 8);
                v8bf l = *(const v8bf*)(srcl + row * KC + col8 * 8);
                *(v8bf*)(Ahi + row * BSTR + col8 * 8) = h;
                *(v8bf*)(Alo + row * BSTR + col8 * 8) = l;
            }
        }
        // Stage B chunk: 128 classes x 32 f32 -> bf16 hi/lo + f32 sumsq.
        #pragma unroll
        for (int j = 0; j < 4; ++j) {
            int i = tid + j * 256;
            int row = i >> 3, col4 = i & 7;
            int gc = cbase + row;
            gc = gc < NUM_CLASSES ? gc : (NUM_CLASSES - 1);   // clamp OOB
            const float* gp = W + (size_t)gc * DIM + kc + col4 * 4;
            v4f v = *(const v4f*)gp;
            if (kc + KC < DIM) __builtin_prefetch(gp + KC, 0, 3);
            v4bf h, l;
            #pragma unroll
            for (int e = 0; e < 4; ++e) {
                float x = v[e];
                ssq[j] = fmaf(x, x, ssq[j]);
                __bf16 hh = (__bf16)x;
                h[e] = hh;
                l[e] = (__bf16)(x - (float)hh);
            }
            *(v4bf*)(Bhi + row * BSTR + col4 * 4) = h;
            *(v4bf*)(Blo + row * BSTR + col4 * 4) = l;
        }
        __syncthreads();

        // One K=32 bf16 WMMA step: dot = Ahi*Bhi + Ahi*Blo + Alo*Bhi.
        // A-fragment loads software-pipelined one m-tile ahead.
        v16bf bh = load_frag(Bhi + (wave * 16 + l15) * BSTR, half);
        v16bf bl = load_frag(Blo + (wave * 16 + l15) * BSTR, half);
        v16bf ah = load_frag(Ahi + l15 * BSTR, half);
        v16bf al = load_frag(Alo + l15 * BSTR, half);
        #pragma unroll
        for (int mt = 0; mt < 16; ++mt) {
            v16bf ah_n, al_n;
            if (mt < 15) {
                ah_n = load_frag(Ahi + ((mt + 1) * 16 + l15) * BSTR, half);
                al_n = load_frag(Alo + ((mt + 1) * 16 + l15) * BSTR, half);
            }
            acc[mt] = __builtin_amdgcn_wmma_f32_16x16x32_bf16(
                false, ah, false, bh, (short)0, acc[mt], false, false);
            acc[mt] = __builtin_amdgcn_wmma_f32_16x16x32_bf16(
                false, ah, false, bl, (short)0, acc[mt], false, false);
            acc[mt] = __builtin_amdgcn_wmma_f32_16x16x32_bf16(
                false, al, false, bh, (short)0, acc[mt], false, false);
            ah = ah_n;
            al = al_n;
        }
    }

    __syncthreads();
    // Merge per-thread sumsq partials: thread covers rows (tid>>3) + j*32.
    #pragma unroll
    for (int j = 0; j < 4; ++j)
        atomicAdd(&wnorm[(tid >> 3) + j * 32], ssq[j]);
    __syncthreads();

    // Epilogue: cosine -> (margin-adjusted) logits -> per-wave LSE partials.
    const int  ncls  = wave * 16 + l15;
    const int  gcls  = cbase + ncls;
    const bool valid = gcls < NUM_CLASSES;
    const float inv  = 1.0f / fmaxf(sqrtf(wnorm[ncls]), EPS_N);

    #pragma unroll
    for (int mt = 0; mt < 16; ++mt) {
        #pragma unroll
        for (int r = 0; r < 8; ++r) {
            int m = mt * 16 + half * 8 + r;
            float cosine = acc[mt][r] * inv;
            float logit;
            if (valid) {
                logit = SCALE_F * cosine;
                if (gcls == sIdty[m]) {
                    float t = cosine;
                    float s2 = fminf(fmaxf(1.0f - t * t, 0.0f), 1.0f);
                    float phi = t * COS_M - sqrtf(s2) * SIN_M;
                    phi = (t > 0.0f) ? phi : t;
                    logit = SCALE_F * phi;
                    tgtl[m] = logit;
                }
            } else {
                logit = -INFINITY;
            }
            float mx = logit;
            #pragma unroll
            for (int msk = 1; msk < 16; msk <<= 1)
                mx = fmaxf(mx, __shfl_xor(mx, msk, 32));
            float ex = valid ? __expf(logit - mx) : 0.0f;
            #pragma unroll
            for (int msk = 1; msk < 16; msk <<= 1)
                ex += __shfl_xor(ex, msk, 32);
            if (l15 == 0) {
                redmax[m * 8 + wave] = mx;
                redsum[m * 8 + wave] = ex;
            }
        }
    }
    __syncthreads();

    // Merge 8 wave partials per row; emit (max, sumexp) for this chunk.
    {
        int m = tid;
        float M = -INFINITY, S = 0.0f;
        #pragma unroll
        for (int w = 0; w < 8; ++w) {
            float m2 = redmax[m * 8 + w];
            float s2 = redsum[m * 8 + w];
            if (m2 > -INFINITY) {
                if (m2 > M) { S = S * __expf(M - m2) + s2; M = m2; }
                else        { S = S + s2 * __expf(m2 - M); }
            }
        }
        pmax[(size_t)m * NUM_CHUNKS + chunk] = M;
        psum[(size_t)m * NUM_CHUNKS + chunk] = S;
    }
}

// ---------------------------------------------------------------------------
// Kernel 3: per-row merge over 2815 chunks -> per-row loss.
// ---------------------------------------------------------------------------
__device__ __forceinline__ void lse_merge(float& M, float& S, float m2, float s2) {
    if (m2 > -INFINITY) {
        if (m2 > M) { S = S * __expf(M - m2) + s2; M = m2; }
        else        { S = S + s2 * __expf(m2 - M); }
    }
}

__global__ void __launch_bounds__(256) row_reduce_kernel(
    const float* __restrict__ pmax, const float* __restrict__ psum,
    const float* __restrict__ tgtl, float* __restrict__ lossv) {
    const int b = blockIdx.x;
    const int t = threadIdx.x;
    float M = -INFINITY, S = 0.0f;
    for (int c = t; c < NUM_CHUNKS; c += 256)
        lse_merge(M, S, pmax[(size_t)b * NUM_CHUNKS + c],
                        psum[(size_t)b * NUM_CHUNKS + c]);
    __shared__ float sm[256], ss[256];
    sm[t] = M; ss[t] = S;
    __syncthreads();
    for (int off = 128; off > 0; off >>= 1) {
        if (t < off) {
            float Ma = sm[t], Sa = ss[t];
            lse_merge(Ma, Sa, sm[t + off], ss[t + off]);
            sm[t] = Ma; ss[t] = Sa;
        }
        __syncthreads();
    }
    if (t == 0) lossv[b] = (sm[0] + logf(ss[0])) - tgtl[b];
}

// ---------------------------------------------------------------------------
// Kernel 4: mean over 256 rows -> scalar loss.
// ---------------------------------------------------------------------------
__global__ void __launch_bounds__(256) mean_kernel(
    const float* __restrict__ lossv, float* __restrict__ out) {
    const int t = threadIdx.x;
    __shared__ float red[256];
    red[t] = lossv[t];
    __syncthreads();
    for (int off = 128; off > 0; off >>= 1) {
        if (t < off) red[t] += red[t + off];
        __syncthreads();
    }
    if (t == 0) out[0] = red[0] / (float)BATCH;
}

// ---------------------------------------------------------------------------
extern "C" void kernel_launch(void* const* d_in, const int* in_sizes, int n_in,
                              void* d_out, int out_size, void* d_ws, size_t ws_size,
                              hipStream_t stream) {
    const float* embs  = (const float*)d_in[0];   // [256, 512]
    const float* W     = (const float*)d_in[1];   // [360232, 512]
    const int*   idtys = (const int*)d_in[2];     // [256]

    // Workspace layout (bytes):
    char* wsb = (char*)d_ws;
    __bf16* AhiG = (__bf16*)wsb;                          // 256*512*2 = 262144 B
    __bf16* AloG = (__bf16*)(wsb + 262144);               // 262144 B
    float*  pmax = (float*)(wsb + 524288);                // 256*2815 f32
    float*  psum = pmax + (size_t)BATCH * NUM_CHUNKS;
    float*  tgtl = psum + (size_t)BATCH * NUM_CHUNKS;     // 256
    float*  lossv = tgtl + BATCH;                         // 256

    norm_split_kernel<<<BATCH, 256, 0, stream>>>(embs, AhiG, AloG);

    const size_t smem_bytes = 61952 + 256 * sizeof(int);  // 62,976 B
    arcface_gemm_kernel<<<NUM_CHUNKS, 256, smem_bytes, stream>>>(
        AhiG, AloG, W, idtys, pmax, psum, tgtl);

    row_reduce_kernel<<<BATCH, 256, 0, stream>>>(pmax, psum, tgtl, lossv);
    mean_kernel<<<1, 256, 0, stream>>>(lossv, (float*)d_out);
}
